// ScaledDotProductAttention_8349416423519
// MI455X (gfx1250) — matrix operand order repaired and checked by reading, not compiled, not run on hardware
//
#include <hip/hip_runtime.h>

typedef __attribute__((ext_vector_type(16))) _Float16 v16h;
typedef __attribute__((ext_vector_type(8)))  float    v8f;

#define SDIM  2048
#define DDIM  64
#define QTILE 128   // queries per block (8 waves x 16)
#define KTILE 32    // keys per inner iteration
#define NWAVE 8
#define KROW  72    // K tile LDS row stride (halves): 144B rows, 16B-aligned, bank-spread
#define VTROW 40    // V^T tile LDS row stride (halves): 80B rows, 16B-aligned, bank-spread
#define PROW  40    // P staging LDS row stride (halves): 80B rows, 16B-aligned, bank-spread

__device__ __forceinline__ float bperm_f(int idx, float v) {
  return __int_as_float(__builtin_amdgcn_ds_bpermute(idx, __float_as_int(v)));
}

__global__ __launch_bounds__(256) void fa_causal_f16wmma(
    const float* __restrict__ qg,
    const float* __restrict__ kg,
    const float* __restrict__ vg,
    float* __restrict__ outg)
{
  __shared__ _Float16 Ks[2][KTILE * KROW];       // K tile, [key][d] f16 (double buffered)
  __shared__ _Float16 Vt[2][DDIM * VTROW];       // V tile TRANSPOSED, [d][key] f16 (double buffered)
  __shared__ _Float16 Ps[NWAVE * 16 * PROW];     // per-wave P staging, [m][n] f16

  const int bh   = blockIdx.x >> 4;              // 64 (b,h) heads
  const int qblk = blockIdx.x & 15;              // 16 query blocks of 128
  const int qb   = qblk * QTILE;
  const size_t head = (size_t)bh * SDIM * DDIM;

  const int tid  = threadIdx.x;
  const int w    = tid >> 5;                     // wave id 0..7
  const int lane = tid & 31;
  const int l16  = lane & 15;
  const int hgrp = lane >> 4;                    // which 16-lane half

  // scale folded into Q; softmax runs in exp2 domain: qscale = (1/sqrt(64)) * log2(e)
  const float qscale = 0.125f * 1.44269504088896340736f;

  // precomputed xor-butterfly byte indices for ds_bpermute (loop invariant)
  int pidx[4];
  #pragma unroll
  for (int j = 0; j < 4; ++j) pidx[j] = ((lane ^ (1 << j)) & 31) << 2;

  // K cooperative-load coordinates: thread -> 8 contiguous d of one key row (2x float4)
  const int kkey = tid >> 3;                     // 0..31
  const int kd0  = (tid & 7) << 3;               // 0..56
  const size_t gK = head + (size_t)kkey * DDIM + kd0;

  // V cooperative-load coordinates (transposed store): thread -> one d column, 8 keys
  const int vd    = tid & 63;                    // 0..63 (consecutive lanes -> coalesced)
  const int vkey0 = (tid >> 6) << 3;             // 0,8,16,24
  const size_t gV = head + (size_t)vkey0 * DDIM + vd;

  const int nkt   = qblk * 4 + 4;                // causal: uniform trip count for the block
  const int qmaxw = qb + w * 16 + 15;            // last query row this wave owns
  const int qminw = qb + w * 16;                 // first query row this wave owns

  // ---- prologue: tile 0 into buffer 0
  {
    const float4* kp = (const float4*)(kg + gK);
    const float4 ka = kp[0], kb = kp[1];
    _Float16* dK = &Ks[0][kkey * KROW + kd0];    // 16B aligned -> ds_store_b128
    dK[0]=(_Float16)ka.x; dK[1]=(_Float16)ka.y; dK[2]=(_Float16)ka.z; dK[3]=(_Float16)ka.w;
    dK[4]=(_Float16)kb.x; dK[5]=(_Float16)kb.y; dK[6]=(_Float16)kb.z; dK[7]=(_Float16)kb.w;
    _Float16* dV = &Vt[0][vd * VTROW + vkey0];   // 16B aligned -> ds_store_b128
    #pragma unroll
    for (int j = 0; j < 8; ++j) dV[j] = (_Float16)vg[gV + (size_t)j * DDIM];
  }

  // ---- Q fragments (scale pre-folded): A-matrix 16x32 f16, contiguous k = 16*hgrp + s (same as K B-frag), two D-chunks.
  const int qrow = qb + w * 16 + l16;
  const float* qrp = qg + head + (size_t)qrow * DDIM;
  v16h aq0, aq1;
  #pragma unroll
  for (int i = 0; i < 8; ++i) {
    const int kidx = 2 * i + 16 * hgrp;
    aq0[2*i]   = (_Float16)(qrp[kidx]          * qscale);
    aq0[2*i+1] = (_Float16)(qrp[kidx + 1]      * qscale);
    aq1[2*i]   = (_Float16)(qrp[32 + kidx]     * qscale);
    aq1[2*i+1] = (_Float16)(qrp[32 + kidx + 1] * qscale);
  }

  // ---- online-softmax state; C/D layout: VGPR r, lane -> (M = r + 8*hgrp, N = l16)
  float mrow[8], lrow[8];
  v8f o0 = {}, o1 = {}, o2 = {}, o3 = {};
  #pragma unroll
  for (int r = 0; r < 8; ++r) { mrow[r] = -1e30f; lrow[r] = 0.0f; }

  __syncthreads();

  for (int kt = 0; kt < nkt; ++kt) {
    const int buf = kt & 1;
    const bool haveNext = (kt + 1 < nkt);

    // ---- issue next tile's global loads now; convert+store after compute (latency hidden by WMMA)
    float4 nka, nkb;
    float nv[8];
    if (haveNext) {
      const size_t off = (size_t)(kt + 1) * KTILE * DDIM;
      const float4* kp = (const float4*)(kg + gK + off);
      nka = kp[0]; nkb = kp[1];
      #pragma unroll
      for (int j = 0; j < 8; ++j) nv[j] = vg[gV + off + (size_t)j * DDIM];
      if (kt + 2 < nkt) {                        // gfx1250 global_prefetch_b8, two tiles ahead
        __builtin_prefetch(kg + gK + off + KTILE * DDIM, 0, 1);
        __builtin_prefetch(vg + gK + off + KTILE * DDIM, 0, 1);
      }
    }

    if (kt * KTILE <= qmaxw) {                   // wave-uniform causal skip
      const _Float16* Kb  = Ks[buf];
      const _Float16* Vtb = Vt[buf];

      // ---- S = Q * K^T : B frag: lane N = l16 -> key col; VGPR i K = 2i + 16*hgrp -> d (pairs contiguous)
      auto loadKB = [&](int ncol, int dchunk) -> v16h {
        v16h b;
        const _Float16* src = &Kb[(ncol * 16 + l16) * KROW + 16 * hgrp + 32 * dchunk];
        #pragma unroll
        for (int i = 0; i < 16; ++i) b[i] = src[i];   // 32B contiguous -> 2x ds_load_b128
        return b;
      };
      v8f s0 = {}, s1 = {};
      s0 = __builtin_amdgcn_wmma_f32_16x16x32_f16(false, aq0, false, loadKB(0, 0), (short)0, s0, false, false);
      s0 = __builtin_amdgcn_wmma_f32_16x16x32_f16(false, aq1, false, loadKB(0, 1), (short)0, s0, false, false);
      s1 = __builtin_amdgcn_wmma_f32_16x16x32_f16(false, aq0, false, loadKB(1, 0), (short)0, s1, false, false);
      s1 = __builtin_amdgcn_wmma_f32_16x16x32_f16(false, aq1, false, loadKB(1, 1), (short)0, s1, false, false);

      // ---- causal mask: only the diagonal tile of this wave needs element masking
      if (kt * KTILE + (KTILE - 1) > qminw) {
        #pragma unroll
        for (int r = 0; r < 8; ++r) {
          const int qi  = qminw + r + 8 * hgrp;
          const int ki0 = kt * KTILE + l16;
          s0[r] = (ki0      <= qi) ? s0[r] : -1e30f;
          s1[r] = (ki0 + 16 <= qi) ? s1[r] : -1e30f;
        }
      }

      // ---- online softmax in exp2 domain (scores already carry log2e/sqrt(d))
      float p0[8], p1[8], fac[8];
      #pragma unroll
      for (int r = 0; r < 8; ++r) {
        float rm = fmaxf(s0[r], s1[r]);
        #pragma unroll
        for (int j = 0; j < 4; ++j) rm = fmaxf(rm, bperm_f(pidx[j], rm));
        const float mn = fmaxf(mrow[r], rm);
        const float f  = exp2f(mrow[r] - mn);
        const float e0 = exp2f(s0[r] - mn);
        const float e1 = exp2f(s1[r] - mn);
        float rs = e0 + e1;
        #pragma unroll
        for (int j = 0; j < 4; ++j) rs += bperm_f(pidx[j], rs);
        lrow[r] = lrow[r] * f + rs;
        mrow[r] = mn;
        fac[r] = f; p0[r] = e0; p1[r] = e1;
      }
      #pragma unroll
      for (int r = 0; r < 8; ++r) {
        o0[r] *= fac[r]; o1[r] *= fac[r]; o2[r] *= fac[r]; o3[r] *= fac[r];
      }

      // ---- P: C-layout -> A-layout via per-wave LDS staging (DS ops in-order per wave)
      _Float16* Pw = &Ps[w * 16 * PROW];
      #pragma unroll
      for (int r = 0; r < 8; ++r) {
        const int m = r + 8 * hgrp;
        Pw[m * PROW + l16]      = (_Float16)p0[r];
        Pw[m * PROW + 16 + l16] = (_Float16)p1[r];
      }
      asm volatile("s_wait_dscnt 0x0" ::: "memory");
      v16h ap;
      #pragma unroll
      for (int i = 0; i < 8; ++i) {
        const int kidx = 2 * i + 16 * hgrp;
        ap[2*i]   = Pw[l16 * PROW + kidx];
        ap[2*i+1] = Pw[l16 * PROW + kidx + 1];
      }

      // ---- O += P * V : B frag from V^T: lane N = l16 -> d col; keys contiguous -> 2x ds_load_b128
      auto loadVB = [&](int j) -> v16h {
        v16h b;
        const _Float16* src = &Vtb[(j * 16 + l16) * VTROW + 16 * hgrp];
        #pragma unroll
        for (int i = 0; i < 16; ++i) b[i] = src[i];   // b[2i],b[2i+1] = keys 2i,2i+1 (+16*hgrp)
        return b;
      };
      o0 = __builtin_amdgcn_wmma_f32_16x16x32_f16(false, ap, false, loadVB(0), (short)0, o0, false, false);
      o1 = __builtin_amdgcn_wmma_f32_16x16x32_f16(false, ap, false, loadVB(1), (short)0, o1, false, false);
      o2 = __builtin_amdgcn_wmma_f32_16x16x32_f16(false, ap, false, loadVB(2), (short)0, o2, false, false);
      o3 = __builtin_amdgcn_wmma_f32_16x16x32_f16(false, ap, false, loadVB(3), (short)0, o3, false, false);
    }

    // ---- convert + store next tile into the other buffer (global-load wait lands here)
    if (haveNext) {
      _Float16* dK = &Ks[buf ^ 1][kkey * KROW + kd0];
      dK[0]=(_Float16)nka.x; dK[1]=(_Float16)nka.y; dK[2]=(_Float16)nka.z; dK[3]=(_Float16)nka.w;
      dK[4]=(_Float16)nkb.x; dK[5]=(_Float16)nkb.y; dK[6]=(_Float16)nkb.z; dK[7]=(_Float16)nkb.w;
      _Float16* dV = &Vt[buf ^ 1][vd * VTROW + vkey0];
      #pragma unroll
      for (int j = 0; j < 8; ++j) dV[j] = (_Float16)nv[j];
    }
    __syncthreads();  // single barrier: publishes buf^1, retires reads of buf
  }

  // ---- epilogue: O / l, write fp32 output (same layout as q)
  float* op = outg + head;
  #pragma unroll
  for (int r = 0; r < 8; ++r) {
    const int m = r + 8 * hgrp;
    const float inv = 1.0f / lrow[r];
    const size_t row = (size_t)(qb + w * 16 + m) * DDIM;
    op[row + l16]      = o0[r] * inv;
    op[row + 16 + l16] = o1[r] * inv;
    op[row + 32 + l16] = o2[r] * inv;
    op[row + 48 + l16] = o3[r] * inv;
  }
}

extern "C" void kernel_launch(void* const* d_in, const int* in_sizes, int n_in,
                              void* d_out, int out_size, void* d_ws, size_t ws_size,
                              hipStream_t stream) {
  const float* q = (const float*)d_in[0];
  const float* k = (const float*)d_in[1];
  const float* v = (const float*)d_in[2];
  // d_in[3] = causal mask; known statically, not read.
  float* out = (float*)d_out;
  dim3 grid(4 * 16 * (SDIM / QTILE));  // 1024 blocks = (b,h) x query-block
  dim3 block(256);                     // 8 wave32 per block
  hipLaunchKernelGGL(fa_causal_f16wmma, grid, block, 0, stream, q, k, v, out);
}
